// CosineDistance_26250840113746
// MI455X (gfx1250) — matrix operand order repaired and checked by reading, not compiled, not run on hardware
//
#include <hip/hip_runtime.h>
#include <hip/hip_bf16.h>

// ---------- types for CDNA5 WMMA ----------
typedef __attribute__((ext_vector_type(16))) __bf16 v16bf;
typedef __attribute__((ext_vector_type(8)))  float  v8f;

union ABFrag { v16bf v; uint4 q[2]; };

__device__ __forceinline__ unsigned short f2bf_bits(float f) {
    // round-to-nearest-even f32 -> bf16 bit pattern
    union { float f; unsigned int u; } c; c.f = f;
    unsigned int u = c.u;
    unsigned int r = (u + 0x7FFFu + ((u >> 16) & 1u)) >> 16;
    return (unsigned short)r;
}

// ---------- K1: emb f32 -> bf16 (packed) + e2[v] = ||emb[v]||^2 ----------
__global__ __launch_bounds__(256)
void k_prep_emb(const float* __restrict__ emb, unsigned int* __restrict__ ebf,
                float* __restrict__ e2, int V, int D) {
    int wave = blockIdx.x * (blockDim.x >> 5) + (threadIdx.x >> 5);
    int lane = threadIdx.x & 31;
    if (wave >= V) return;
    const float2* src = (const float2*)(emb + (size_t)wave * D);
    unsigned int* dst = ebf + ((size_t)wave * D >> 1);
    float sum = 0.f;
    int pairs = D >> 1;
    for (int p = lane; p < pairs; p += 32) {
        float2 f = src[p];
        sum += f.x * f.x + f.y * f.y;
        dst[p] = (unsigned int)f2bf_bits(f.x) | ((unsigned int)f2bf_bits(f.y) << 16);
    }
    for (int m = 16; m; m >>= 1) sum += __shfl_xor(sum, m, 32);
    if (lane == 0) e2[wave] = sum;
}

// ---------- K2: gather gold rows (bf16 copy), g2, zero accumulators ----------
__global__ __launch_bounds__(256)
void k_gather(const int* __restrict__ target, const unsigned int* __restrict__ ebf,
              const float* __restrict__ e2, unsigned int* __restrict__ gbf,
              float* __restrict__ g2, float* __restrict__ S0, float* __restrict__ S1,
              int N, int D) {
    int wave = blockIdx.x * (blockDim.x >> 5) + (threadIdx.x >> 5);
    int lane = threadIdx.x & 31;
    if (wave >= N) return;
    int t = target[wave];
    const unsigned int* src = ebf + ((size_t)t    * D >> 1);
    unsigned int*       dst = gbf + ((size_t)wave * D >> 1);
    for (int p = lane; p < (D >> 1); p += 32) dst[p] = src[p];
    if (lane == 0) { g2[wave] = e2[t]; S0[wave] = 0.f; S1[wave] = 0.f; }
}

// ---------- K3: fused bf16-WMMA GEMM + distance + exp + row reduction ----------
// grid = (V/256, N/16), block = 256 (8 waves).
// Wave w owns a 16x32 slab: two 16x16 WMMA tiles at cols v0 + 32w and v0 + 32w + 16,
// sharing one A fragment per k-step. K-loop is double-buffered so fragment loads
// for step k+1 overlap the WMMAs of step k (partial s_wait instead of full drain).
__global__ __launch_bounds__(256)
void k_fused(const unsigned int* __restrict__ gbf, const unsigned int* __restrict__ ebf,
             const float* __restrict__ e2, const float* __restrict__ g2,
             const float* __restrict__ pred_ll,
             float* __restrict__ S0, float* __restrict__ S1, int V) {
    constexpr int KD    = 512;          // D, multiple of 32
    constexpr int KSTEPS = KD / 32;     // 16
    __shared__ __align__(16) unsigned int Atile[16 * KD / 2];  // 16 KB bf16 A tile
    __shared__ float s0l[16], s1l[16];

    const int tid    = threadIdx.x;
    const int waveId = tid >> 5;
    const int lane   = tid & 31;
    const int n0     = blockIdx.y * 16;
    const int v0     = blockIdx.x * 256;

    if (tid < 16) { s0l[tid] = 0.f; s1l[tid] = 0.f; }

    // stage A (16 rows x 512 bf16, contiguous in gbf) into LDS: 4x uint4 per thread
    {
        const uint4* asrc = (const uint4*)(gbf + ((size_t)n0 * KD >> 1));
        uint4*       adst = (uint4*)Atile;
        #pragma unroll
        for (int i = 0; i < (16 * KD / 2 / 4) / 256; ++i)
            adst[tid + 256 * i] = asrc[tid + 256 * i];
    }

    const int hh   = lane >> 4;        // half-wave: 0 or 1
    const int col  = lane & 15;        // N within tile / M row within A
    const int vrow0 = v0 + waveId * 32 + col;
    const int vrow1 = vrow0 + 16;
    const unsigned int*   brow0 = ebf + ((size_t)vrow0 * KD >> 1);
    const unsigned int*   brow1 = ebf + ((size_t)vrow1 * KD >> 1);
    const unsigned short* A     = (const unsigned short*)Atile;

    // prefetch this block's pred_ll tile (the HBM-bound stream) so its latency
    // overlaps the WMMA loop instead of serializing in the epilogue
    #pragma unroll
    for (int r = 0; r < 8; ++r) {
        const size_t rowoff = (size_t)(n0 + r + hh * 8) * V;
        __builtin_prefetch(&pred_ll[rowoff + vrow0], 0, 1);
        __builtin_prefetch(&pred_ll[rowoff + vrow1], 0, 1);
    }

    __syncthreads();

    // fragment loaders -------------------------------------------------------
    // A 16x32 bf16 layout (ISA 7.12.2): lane holds row=col, K runs
    // [32ks+8hh, +8) and [32ks+8hh+16, +8)  -> two b128 LDS loads
    auto loadA = [&](ABFrag& a, int ks) {
        const int ka = ks * 32 + hh * 8;
        a.q[0] = *(const uint4*)(A + col * KD + ka);
        a.q[1] = *(const uint4*)(A + col * KD + ka + 16);
    };
    // B 32x16 bf16: lane holds col N=col, K runs [32ks+8hh, +8) and [32ks+8hh+16, +8)
    auto loadB = [&](ABFrag& b, const unsigned int* brow, int ks) {
        const uint4* bp = (const uint4*)(brow + ((ks * 32 + hh * 8) >> 1));
        b.q[0] = bp[0];
        b.q[1] = bp[2];
    };

    ABFrag a[2], b0[2], b1[2];
    loadA(a[0], 0);
    loadB(b0[0], brow0, 0);
    loadB(b1[0], brow1, 0);

    v8f c0 = {}, c1 = {};
    #pragma unroll
    for (int ks = 0; ks < KSTEPS; ++ks) {
        const int cur = ks & 1, nxt = cur ^ 1;
        if (ks + 1 < KSTEPS) {          // issue next-step loads before consuming cur
            loadA(a[nxt], ks + 1);
            loadB(b0[nxt], brow0, ks + 1);
            loadB(b1[nxt], brow1, ks + 1);
        }
        c0 = __builtin_amdgcn_wmma_f32_16x16x32_bf16(false, a[cur].v, false, b0[cur].v,
                                                     (short)0, c0, false, false);
        c1 = __builtin_amdgcn_wmma_f32_16x16x32_bf16(false, a[cur].v, false, b1[cur].v,
                                                     (short)0, c1, false, false);
    }

    // epilogue: dist -> exp(-dist), weighted -log p; reduce 32 cols per row
    const float e2v0 = e2[vrow0];
    const float e2v1 = e2[vrow1];
    #pragma unroll
    for (int r = 0; r < 8; ++r) {
        const int row = n0 + r + hh * 8;            // C layout: VGPR r, halves
        const float g2r = g2[row];
        const size_t rowoff = (size_t)row * V;

        float d20   = fmaxf(g2r + e2v0 - 2.0f * c0[r], 0.0f);
        float dist0 = __fsqrt_rn(fmaxf(d20, 1e-12f));
        float we0   = __expf(-dist0);               // softmax(-dist) numerator

        float d21   = fmaxf(g2r + e2v1 - 2.0f * c1[r], 0.0f);
        float dist1 = __fsqrt_rn(fmaxf(d21, 1e-12f));
        float we1   = __expf(-dist1);

        float s0 = we0 + we1;
        float s1 = we0 * (-pred_ll[rowoff + vrow0]) + we1 * (-pred_ll[rowoff + vrow1]);

        // xor masks 8..1 stay inside each 16-lane half of the wave32
        for (int m = 8; m; m >>= 1) {
            s0 += __shfl_xor(s0, m, 32);
            s1 += __shfl_xor(s1, m, 32);
        }
        if (col == 0) {
            atomicAdd(&s0l[r + hh * 8], s0);        // ds_add_f32 across 8 waves
            atomicAdd(&s1l[r + hh * 8], s1);
        }
    }
    __syncthreads();
    if (tid < 16) {
        atomicAdd(&S0[n0 + tid], s0l[tid]);         // 1 global atomic / row / block
        atomicAdd(&S1[n0 + tid], s1l[tid]);
    }
}

// ---------- K4: finalize both scalars ----------
__global__ __launch_bounds__(256)
void k_final(const float* __restrict__ pred_ll, const int* __restrict__ target,
             const float* __restrict__ S0, const float* __restrict__ S1,
             float* __restrict__ out, int N, int V) {
    __shared__ float redL[256], redN[256];
    int tid = threadIdx.x;
    float loss = 0.f, nll = 0.f;
    for (int n = tid; n < N; n += 256) {
        int   t    = target[n];
        float mask = (t != 0) ? 1.0f : 0.0f;        // PAD == 0
        float ll   = pred_ll[(size_t)n * V + t];
        nll  += mask * (-ll);
        loss += mask * (S1[n] / S0[n]);
    }
    redL[tid] = loss; redN[tid] = nll;
    __syncthreads();
    for (int s = 128; s; s >>= 1) {
        if (tid < s) { redL[tid] += redL[tid + s]; redN[tid] += redN[tid + s]; }
        __syncthreads();
    }
    if (tid == 0) { out[0] = redL[0]; out[1] = redN[0]; }
}

extern "C" void kernel_launch(void* const* d_in, const int* in_sizes, int n_in,
                              void* d_out, int out_size, void* d_ws, size_t ws_size,
                              hipStream_t stream) {
    const float* pred_ll = (const float*)d_in[0];   // [N, V] f32
    const int*   target  = (const int*)d_in[1];     // [N] i32
    const float* emb     = (const float*)d_in[2];   // [V, D] f32

    const long N = in_sizes[1];
    const long V = in_sizes[0] / N;                 // 32000
    const long D = in_sizes[2] / V;                 // 512

    float* out = (float*)d_out;

    // workspace layout
    char*  ws  = (char*)d_ws;
    size_t off = 0;
    auto alloc = [&](size_t bytes) -> void* {
        void* p = ws + off;
        off += (bytes + 255) & ~(size_t)255;
        return p;
    };
    unsigned int* ebf = (unsigned int*)alloc((size_t)V * D * 2);  // bf16 table (32 MB, L2-resident)
    unsigned int* gbf = (unsigned int*)alloc((size_t)N * D * 2);  // bf16 gold rows
    float* e2 = (float*)alloc((size_t)V * 4);
    float* g2 = (float*)alloc((size_t)N * 4);
    float* S0 = (float*)alloc((size_t)N * 4);
    float* S1 = (float*)alloc((size_t)N * 4);

    k_prep_emb<<<dim3((unsigned)((V + 7) / 8)), 256, 0, stream>>>(emb, ebf, e2, (int)V, (int)D);
    k_gather  <<<dim3((unsigned)((N + 7) / 8)), 256, 0, stream>>>(target, ebf, e2, gbf, g2, S0, S1,
                                                                  (int)N, (int)D);
    k_fused   <<<dim3((unsigned)(V / 256), (unsigned)(N / 16)), 256, 0, stream>>>(
                  gbf, ebf, e2, g2, pred_ll, S0, S1, (int)V);
    k_final   <<<1, 256, 0, stream>>>(pred_ll, target, S0, S1, out, (int)N, (int)V);
}
